// CausalAttention_ov1_57518202028180
// MI455X (gfx1250) — matrix operand order repaired and checked by reading, not compiled, not run on hardware
//
#include <hip/hip_runtime.h>
#include <stdint.h>

typedef __attribute__((ext_vector_type(16))) __bf16 v16bf;
typedef __attribute__((ext_vector_type(8)))  float  v8f;

constexpr int S_SEQ = 4096;
constexpr int D_DIM = 2048;
constexpr int LDT   = 72;          // LDS row stride (bf16 elems): 64 + 8 pad, 144B (16B multiple)
constexpr int TILE_ELEMS = 128 * LDT;   // one LDS buffer (bf16 elems)
constexpr float NEG = -3.0e38f;

__device__ __forceinline__ uint16_t f2bf(float f) {
  union { float f; uint32_t u; } c; c.f = f;
  uint32_t u = c.u;
  u += 0x7FFFu + ((u >> 16) & 1u);   // round-to-nearest-even
  return (uint16_t)(u >> 16);
}

// ---- CDNA5 async global->LDS copy (ASYNCcnt). INST_OFFSET applies to BOTH addresses. ----
__device__ __forceinline__ void async_cp_row64(const uint16_t* g, const uint16_t* l) {
  uint64_t ga = (uint64_t)(uintptr_t)g;           // global byte address (VGPR pair)
  uint32_t la = (uint32_t)(uintptr_t)l;           // low 32 bits of flat LDS addr == LDS offset
  asm volatile(
      "global_load_async_to_lds_b128 %0, %1, off\n\t"
      "global_load_async_to_lds_b128 %0, %1, off offset:16\n\t"
      "global_load_async_to_lds_b128 %0, %1, off offset:32\n\t"
      "global_load_async_to_lds_b128 %0, %1, off offset:48"
      :: "v"(la), "v"(ga) : "memory");
}
__device__ __forceinline__ void wait_async0() {
  asm volatile("s_wait_asynccnt 0x0" ::: "memory");
}

// A fragment (16x32 bf16, ISA layout): lane m=L%16, h=L/16.
// VGPR v pairs at col k0 = 16*(v/4) + 8*h + 2*(v%4)  -> cols [8h..8h+7] and [16+8h..16+8h+7]
__device__ __forceinline__ v16bf ld_afrag(const uint16_t* rowp, int h) {
  union { uint4 q[2]; v16bf v; } u;
  u.q[0] = *(const uint4*)(rowp + 8 * h);
  u.q[1] = *(const uint4*)(rowp + 16 + 8 * h);
  return u.v;
}
// B fragment (32x16 bf16): lane n=L%16, h=L/16. same k order as A -> cols [8h..8h+7] and [16+8h..16+8h+7]
__device__ __forceinline__ v16bf ld_bfrag(const uint16_t* rowp, int h) {
  union { uint4 q[2]; v16bf v; } u;
  u.q[0] = *(const uint4*)(rowp + 8 * h);
  u.q[1] = *(const uint4*)(rowp + 16 + 8 * h);
  return u.v;
}

__device__ __forceinline__ v8f wmma_bf16(v16bf a, v16bf b, v8f c) {
  return __builtin_amdgcn_wmma_f32_16x16x32_bf16(false, a, false, b, (short)0, c, false, false);
}

// Double-buffered GEMM k-loop. BK=64 per tile, 2 wmma k-steps, one barrier per tile.
// Pipeline: [wait asynccnt 0] -> barrier -> issue async loads for kt+1 (buf^1) -> wmma on buf.
template <bool TRANSB>
__device__ __forceinline__ void gemm_loop(
    const uint16_t* __restrict__ A, int lda,      // row 0 of the 128-row block
    const uint16_t* __restrict__ B, int ldb,      // !TRANSB: row 0 of 128 N-rows;  TRANSB: col base applied
    int kTiles, uint16_t* ldsA, uint16_t* ldsB,   // each 2*TILE_ELEMS (ping-pong)
    int tid, int wm, int wn, int ln, int h,
    v8f acc[2][4]) {
  const int rA = tid >> 1, cA = (tid & 1) * 32;   // 2 threads/row, 32 elems (64B) each

  auto stageA = [&](int buf, int kt) {
    async_cp_row64(A + (size_t)rA * lda + kt * 64 + cA,
                   ldsA + buf * TILE_ELEMS + rA * LDT + cA);
  };
  auto stageB = [&](int buf, int kt) {
    if constexpr (!TRANSB) {
      async_cp_row64(B + (size_t)rA * ldb + kt * 64 + cA,
                     ldsB + buf * TILE_ELEMS + rA * LDT + cA);
    } else {
      // V tile: 64 (k=j) x 128 (n=o) in memory, stored transposed: ldsB[n][k]
      int j = tid >> 2, o0 = (tid & 3) * 32;
      union { uint4 q[4]; uint16_t e[32]; } t;
      const uint4* g = (const uint4*)(B + (size_t)(kt * 64 + j) * ldb + o0);
      t.q[0] = g[0]; t.q[1] = g[1]; t.q[2] = g[2]; t.q[3] = g[3];
      uint16_t* lb = ldsB + buf * TILE_ELEMS;
#pragma unroll
      for (int e = 0; e < 32; ++e) lb[(o0 + e) * LDT + j] = t.e[e];
    }
  };

  stageA(0, 0);
  stageB(0, 0);
  for (int kt = 0; kt < kTiles; ++kt) {
    int buf = kt & 1;
    wait_async0();
    __syncthreads();
    if (kt + 1 < kTiles) { stageA(buf ^ 1, kt + 1); stageB(buf ^ 1, kt + 1); }
    const uint16_t* bA = ldsA + buf * TILE_ELEMS;
    const uint16_t* bB = ldsB + buf * TILE_ELEMS;
#pragma unroll
    for (int ks = 0; ks < 2; ++ks) {
      int ko = ks * 32;
      v16bf af[2], bfr[4];
#pragma unroll
      for (int mt = 0; mt < 2; ++mt)
        af[mt] = ld_afrag(bA + (wm * 32 + mt * 16 + ln) * LDT + ko, h);
#pragma unroll
      for (int nt = 0; nt < 4; ++nt)
        bfr[nt] = ld_bfrag(bB + (wn * 64 + nt * 16 + ln) * LDT + ko, h);
#pragma unroll
      for (int mt = 0; mt < 2; ++mt)
#pragma unroll
        for (int nt = 0; nt < 4; ++nt)
          acc[mt][nt] = wmma_bf16(af[mt], bfr[nt], acc[mt][nt]);
    }
  }
}

// ---------------- kernels ----------------

__global__ void k_cast(const float* __restrict__ src, uint16_t* __restrict__ dst, int n4) {
  int i = blockIdx.x * blockDim.x + threadIdx.x;
  if (i >= n4) return;
  float4 v = ((const float4*)src)[i];
  uint32_t lo = (uint32_t)f2bf(v.x) | ((uint32_t)f2bf(v.y) << 16);
  uint32_t hi = (uint32_t)f2bf(v.z) | ((uint32_t)f2bf(v.w) << 16);
  ((uint2*)dst)[i] = make_uint2(lo, hi);
}

// Y = Xb @ Wb^T  (W stored [Dout,Din] row-major -> B tile rows are output cols), bf16 out + scale
__global__ __launch_bounds__(256) void k_qkv(const uint16_t* __restrict__ xb,
                                             const uint16_t* __restrict__ wb,
                                             uint16_t* __restrict__ out, float scale) {
  __shared__ __align__(16) uint16_t ldsA[2 * TILE_ELEMS];
  __shared__ __align__(16) uint16_t ldsB[2 * TILE_ELEMS];
  int tid = threadIdx.x, lane = tid & 31, wid = tid >> 5;
  int ln = lane & 15, h = lane >> 4, wm = wid & 3, wn = wid >> 2;
  int nb = blockIdx.x, rb = blockIdx.y;
  const uint16_t* A = xb + (size_t)rb * 128 * D_DIM;
  const uint16_t* B = wb + (size_t)nb * 128 * D_DIM;
  v8f acc[2][4] = {};
  gemm_loop<false>(A, D_DIM, B, D_DIM, D_DIM / 64, ldsA, ldsB, tid, wm, wn, ln, h, acc);
#pragma unroll
  for (int mt = 0; mt < 2; ++mt)
#pragma unroll
    for (int nt = 0; nt < 4; ++nt)
#pragma unroll
      for (int r = 0; r < 8; ++r) {
        int row = rb * 128 + wm * 32 + mt * 16 + r + 8 * h;
        int col = nb * 128 + wn * 64 + nt * 16 + ln;
        out[(size_t)row * D_DIM + col] = f2bf(acc[mt][nt][r] * scale);
      }
}

// per-(128x128 block) causal row stats: block max and sumexp (relative to block max)
__global__ __launch_bounds__(256) void k_sstats(const uint16_t* __restrict__ qb,
                                                const uint16_t* __restrict__ kb,
                                                float* __restrict__ bmax,
                                                float* __restrict__ bsum) {
  int cb = blockIdx.x, rb = blockIdx.y;
  if (cb > rb) return;
  __shared__ __align__(16) uint16_t ldsA[2 * TILE_ELEMS];
  __shared__ __align__(16) uint16_t ldsB[2 * TILE_ELEMS];
  __shared__ float redM[128][2];
  __shared__ float redS[128][2];
  int tid = threadIdx.x, lane = tid & 31, wid = tid >> 5;
  int ln = lane & 15, h = lane >> 4, wm = wid & 3, wn = wid >> 2;
  const uint16_t* A = qb + (size_t)rb * 128 * D_DIM;
  const uint16_t* B = kb + (size_t)cb * 128 * D_DIM;
  v8f acc[2][4] = {};
  gemm_loop<false>(A, D_DIM, B, D_DIM, D_DIM / 64, ldsA, ldsB, tid, wm, wn, ln, h, acc);
  int rowBase = rb * 128, colBase = cb * 128;
#pragma unroll
  for (int mt = 0; mt < 2; ++mt)
#pragma unroll
    for (int r = 0; r < 8; ++r) {
      int gi = rowBase + wm * 32 + mt * 16 + r + 8 * h;
      float mx = NEG;
#pragma unroll
      for (int nt = 0; nt < 4; ++nt) {
        int gj = colBase + wn * 64 + nt * 16 + ln;
        if (gj <= gi) mx = fmaxf(mx, acc[mt][nt][r]);
      }
      for (int o = 8; o; o >>= 1) mx = fmaxf(mx, __shfl_xor(mx, o, 16));
      float sm = 0.f;
#pragma unroll
      for (int nt = 0; nt < 4; ++nt) {
        int gj = colBase + wn * 64 + nt * 16 + ln;
        if (gj <= gi) sm += __expf(acc[mt][nt][r] - mx);
      }
      for (int o = 8; o; o >>= 1) sm += __shfl_xor(sm, o, 16);
      if (ln == 0) {
        int lr = wm * 32 + mt * 16 + r + 8 * h;
        redM[lr][wn] = mx; redS[lr][wn] = sm;
      }
    }
  __syncthreads();
  if (tid < 128) {
    float m1 = redM[tid][0], m2 = redM[tid][1];
    float l1 = redS[tid][0], l2 = redS[tid][1];
    float m = fmaxf(m1, m2);
    float l = l1 * __expf(m1 - m) + l2 * __expf(m2 - m);
    int gr = rb * 128 + tid;
    bmax[(size_t)gr * 32 + cb] = m;
    bsum[(size_t)gr * 32 + cb] = l;
  }
}

__global__ void k_reduce(const float* __restrict__ bmax, const float* __restrict__ bsum,
                         float* __restrict__ rowM, float* __restrict__ rowInvL) {
  int gr = blockIdx.x * blockDim.x + threadIdx.x;
  if (gr >= S_SEQ) return;
  int rb = gr >> 7;
  float m = NEG, l = 0.f;
  for (int cb = 0; cb <= rb; ++cb) {
    float bm = bmax[(size_t)gr * 32 + cb], bl = bsum[(size_t)gr * 32 + cb];
    float mn = fmaxf(m, bm);
    l = l * __expf(m - mn) + bl * __expf(bm - mn);
    m = mn;
  }
  rowM[gr] = m;
  rowInvL[gr] = 1.f / l;
}

// recompute scores, write normalized causal P (bf16)
__global__ __launch_bounds__(256) void k_p(const uint16_t* __restrict__ qb,
                                           const uint16_t* __restrict__ kb,
                                           const float* __restrict__ rowM,
                                           const float* __restrict__ rowInvL,
                                           uint16_t* __restrict__ P) {
  int cb = blockIdx.x, rb = blockIdx.y;
  if (cb > rb) return;
  __shared__ __align__(16) uint16_t ldsA[2 * TILE_ELEMS];
  __shared__ __align__(16) uint16_t ldsB[2 * TILE_ELEMS];
  int tid = threadIdx.x, lane = tid & 31, wid = tid >> 5;
  int ln = lane & 15, h = lane >> 4, wm = wid & 3, wn = wid >> 2;
  const uint16_t* A = qb + (size_t)rb * 128 * D_DIM;
  const uint16_t* B = kb + (size_t)cb * 128 * D_DIM;
  v8f acc[2][4] = {};
  gemm_loop<false>(A, D_DIM, B, D_DIM, D_DIM / 64, ldsA, ldsB, tid, wm, wn, ln, h, acc);
  int rowBase = rb * 128, colBase = cb * 128;
#pragma unroll
  for (int mt = 0; mt < 2; ++mt)
#pragma unroll
    for (int nt = 0; nt < 4; ++nt)
#pragma unroll
      for (int r = 0; r < 8; ++r) {
        int gi = rowBase + wm * 32 + mt * 16 + r + 8 * h;
        int gj = colBase + wn * 64 + nt * 16 + ln;
        float p = 0.f;
        if (gj <= gi) p = __expf(acc[mt][nt][r] - rowM[gi]) * rowInvL[gi];
        P[(size_t)gi * S_SEQ + gj] = f2bf(p);
      }
}

// O = P @ V (fp32 out); skip zero upper-triangle k-blocks; V staged transposed in LDS
__global__ __launch_bounds__(256) void k_out(const uint16_t* __restrict__ P,
                                             const uint16_t* __restrict__ vb,
                                             float* __restrict__ out) {
  __shared__ __align__(16) uint16_t ldsA[2 * TILE_ELEMS];
  __shared__ __align__(16) uint16_t ldsB[2 * TILE_ELEMS];
  int tid = threadIdx.x, lane = tid & 31, wid = tid >> 5;
  int ln = lane & 15, h = lane >> 4, wm = wid & 3, wn = wid >> 2;
  int nb = blockIdx.x, rb = blockIdx.y;
  const uint16_t* A = P + (size_t)rb * 128 * S_SEQ;
  const uint16_t* B = vb + (size_t)nb * 128;        // column base; rows indexed by k inside
  v8f acc[2][4] = {};
  int kts = (rb + 1) * 2;                            // only j <= (rb+1)*128
  gemm_loop<true>(A, S_SEQ, B, D_DIM, kts, ldsA, ldsB, tid, wm, wn, ln, h, acc);
#pragma unroll
  for (int mt = 0; mt < 2; ++mt)
#pragma unroll
    for (int nt = 0; nt < 4; ++nt)
#pragma unroll
      for (int r = 0; r < 8; ++r) {
        int row = rb * 128 + wm * 32 + mt * 16 + r + 8 * h;
        int col = nb * 128 + wn * 64 + nt * 16 + ln;
        out[(size_t)row * D_DIM + col] = acc[mt][nt][r];
      }
}

// ---------------- launch ----------------

extern "C" void kernel_launch(void* const* d_in, const int* in_sizes, int n_in,
                              void* d_out, int out_size, void* d_ws, size_t ws_size,
                              hipStream_t stream) {
  (void)in_sizes; (void)n_in; (void)out_size; (void)ws_size;
  const float* x  = (const float*)d_in[0];
  const float* Wq = (const float*)d_in[1];
  const float* Wk = (const float*)d_in[2];
  const float* Wv = (const float*)d_in[3];

  char* ws = (char*)d_ws;
  size_t off = 0;
  auto carve = [&](size_t bytes) -> void* {
    void* p = ws + off;
    off += (bytes + 255) & ~(size_t)255;
    return p;
  };
  uint16_t* xb   = (uint16_t*)carve((size_t)S_SEQ * D_DIM * 2);
  uint16_t* wqb  = (uint16_t*)carve((size_t)D_DIM * D_DIM * 2);
  uint16_t* wkb  = (uint16_t*)carve((size_t)D_DIM * D_DIM * 2);
  uint16_t* wvb  = (uint16_t*)carve((size_t)D_DIM * D_DIM * 2);
  uint16_t* qb   = (uint16_t*)carve((size_t)S_SEQ * D_DIM * 2);
  uint16_t* kbuf = (uint16_t*)carve((size_t)S_SEQ * D_DIM * 2);
  uint16_t* vbuf = (uint16_t*)carve((size_t)S_SEQ * D_DIM * 2);
  uint16_t* Pm   = (uint16_t*)carve((size_t)S_SEQ * S_SEQ * 2);
  float* bmax    = (float*)carve((size_t)S_SEQ * 32 * 4);
  float* bsum    = (float*)carve((size_t)S_SEQ * 32 * 4);
  float* rowM    = (float*)carve((size_t)S_SEQ * 4);
  float* rowInvL = (float*)carve((size_t)S_SEQ * 4);

  const float qscale = 1.0f / sqrtf((float)D_DIM);   // fold score scale into q

  int n4 = S_SEQ * D_DIM / 4;
  k_cast<<<(n4 + 255) / 256, 256, 0, stream>>>(x, xb, n4);
  n4 = D_DIM * D_DIM / 4;
  k_cast<<<(n4 + 255) / 256, 256, 0, stream>>>(Wq, wqb, n4);
  k_cast<<<(n4 + 255) / 256, 256, 0, stream>>>(Wk, wkb, n4);
  k_cast<<<(n4 + 255) / 256, 256, 0, stream>>>(Wv, wvb, n4);

  dim3 g1(D_DIM / 128, S_SEQ / 128);
  k_qkv<<<g1, 256, 0, stream>>>(xb, wqb, qb, qscale);
  k_qkv<<<g1, 256, 0, stream>>>(xb, wkb, kbuf, 1.0f);
  k_qkv<<<g1, 256, 0, stream>>>(xb, wvb, vbuf, 1.0f);

  dim3 g2(S_SEQ / 128, S_SEQ / 128);
  k_sstats<<<g2, 256, 0, stream>>>(qb, kbuf, bmax, bsum);
  k_reduce<<<(S_SEQ + 255) / 256, 256, 0, stream>>>(bmax, bsum, rowM, rowInvL);
  k_p<<<g2, 256, 0, stream>>>(qb, kbuf, rowM, rowInvL, Pm);

  dim3 g3(D_DIM / 128, S_SEQ / 128);
  k_out<<<g3, 256, 0, stream>>>(Pm, vbuf, (float*)d_out);
}
